// LinearAttentionBlock_35364760715517
// MI455X (gfx1250) — matrix operand order repaired and checked by reading, not compiled, not run on hardware
//
#include <hip/hip_runtime.h>
#include <hip/hip_bf16.h>
#include <math.h>

// Problem dims
#define B_    8
#define T_    512
#define FIN_  256
#define D_    128
#define BT_   4096
#define CHUNK 64
#define NCH   8      // T_/CHUNK
#define DD    16384  // D_*D_

typedef __attribute__((ext_vector_type(16))) __bf16 v16bf;
typedef __attribute__((ext_vector_type(8)))  __bf16 v8bf;
typedef __attribute__((ext_vector_type(4)))  __bf16 v4bf;
typedef __attribute__((ext_vector_type(8)))  float  v8f;
typedef __attribute__((ext_vector_type(4)))  float  v4f;

union V16 { v16bf v; v8bf h[2]; };
union V8B { v8bf v; __bf16 e[8]; };
union V4B { v4bf v; __bf16 e[4]; };

// ---- WMMA fragment loaders (contiguous k order k = 16*(L/16) + j, same for A and B) ----
// A 16x32: lane L, row M=L%16; elems j = K k0 + j, where k0 = kbase + (L/16)*16.
// Two contiguous 16B loads.
__device__ __forceinline__ v16bf ldA(const __bf16* p) {
    V16 u;
    u.h[0] = *(const v8bf*)(p);
    u.h[1] = *(const v8bf*)(p + 8);
    return u.v;
}
// B 32x16: lane L, col N=L%16; elems j = K (L/16)*16 + j  -> 16 contiguous bf16.
__device__ __forceinline__ v16bf ldB(const __bf16* p) {
    V16 u;
    u.h[0] = *(const v8bf*)(p);
    u.h[1] = *(const v8bf*)(p + 8);
    return u.v;
}
#define WMMA_BF16(a, b, c) \
    __builtin_amdgcn_wmma_f32_16x16x32_bf16(false, (a), false, (b), (short)0, (c), false, false)

__device__ __forceinline__ float clip1e20(float x) {
    return fminf(fmaxf(x, -1e20f), 1e20f);
}

// ============ 0. weight transpose + fp32->bf16 (N-major so B-frags are contiguous) ============
__global__ void la_prep_weights(const float* Wk, const float* Wq, const float* Wv,
                                const float* W1, const float* W2, const float* Ws,
                                __bf16* Wkt, __bf16* Wqt, __bf16* Wvt,
                                __bf16* W1t, __bf16* W2t, __bf16* Wst) {
    const float* src; __bf16* dst; int K, N;
    switch (blockIdx.y) {
        case 0: src = Wk; dst = Wkt; K = FIN_; N = D_; break;
        case 1: src = Wq; dst = Wqt; K = FIN_; N = D_; break;
        case 2: src = Wv; dst = Wvt; K = FIN_; N = D_; break;
        case 3: src = Ws; dst = Wst; K = FIN_; N = D_; break;
        case 4: src = W1; dst = W1t; K = D_;  N = D_; break;
        default: src = W2; dst = W2t; K = D_; N = D_; break;
    }
    int idx = blockIdx.x * 256 + threadIdx.x;
    if (idx < K * N) {
        int k = idx / N, n = idx % N;
        dst[(size_t)n * K + k] = (__bf16)src[idx];
    }
}

// ============ 1. LayerNorm (one wave per row, wave32 shfl reduction) ============
__global__ void la_layernorm(const float* __restrict__ x,
                             const float* __restrict__ gamma,
                             const float* __restrict__ beta,
                             __bf16* __restrict__ xn) {
    int row  = blockIdx.x * 8 + (threadIdx.x >> 5);
    int lane = threadIdx.x & 31;
    const float* xr = x + (size_t)row * FIN_;
    float v[8], s = 0.f, s2 = 0.f;
#pragma unroll
    for (int i = 0; i < 8; ++i) {
        float t = xr[lane * 8 + i];
        v[i] = t; s += t; s2 += t * t;
    }
#pragma unroll
    for (int m = 16; m >= 1; m >>= 1) {
        s  += __shfl_xor(s,  m, 32);
        s2 += __shfl_xor(s2, m, 32);
    }
    float mu  = s * (1.0f / FIN_);
    float var = s2 * (1.0f / FIN_) - mu * mu;
    float inv = rsqrtf(var + 1e-5f);
    V8B o;
#pragma unroll
    for (int i = 0; i < 8; ++i) {
        int c = lane * 8 + i;
        o.e[i] = (__bf16)((v[i] - mu) * inv * gamma[c] + beta[c]);
    }
    *(v8bf*)(xn + (size_t)row * FIN_ + lane * 8) = o.v;
}

// ============ 2. fused Q/K/V/residual GEMM: [4096,256]bf16 x [256,128]x4 (WMMA) ============
// Also emits time-major bf16 copies of activated K and V for the chunk-sum WMMAs.
__global__ void la_qkvr_gemm(const __bf16* __restrict__ xn,
                             const __bf16* __restrict__ Wkt, const __bf16* __restrict__ Wqt,
                             const __bf16* __restrict__ Wvt, const __bf16* __restrict__ Wst,
                             const float* __restrict__ bs,
                             float* __restrict__ Qf, float* __restrict__ Kf,
                             float* __restrict__ Vf, float* __restrict__ Rf,
                             __bf16* __restrict__ Ktb, __bf16* __restrict__ Vtb) {
    int lane = threadIdx.x & 31;
    int tile = blockIdx.x * 4 + (threadIdx.x >> 5);   // 0..2047
    int m0 = (tile >> 3) * 16, n0 = (tile & 7) * 16;
    int hs = lane >> 4;
    int rowA = m0 + (lane & 15);
    int colB = n0 + (lane & 15);
    v8f cq = {}, ck = {}, cv = {}, cr = {};
#pragma unroll
    for (int kk = 0; kk < FIN_; kk += 32) {
        v16bf a = ldA(xn + (size_t)rowA * FIN_ + kk + hs * 16);
        int kb = kk + hs * 16;
        v16bf bq = ldB(Wqt + (size_t)colB * FIN_ + kb);
        v16bf bk = ldB(Wkt + (size_t)colB * FIN_ + kb);
        v16bf bv = ldB(Wvt + (size_t)colB * FIN_ + kb);
        v16bf br = ldB(Wst + (size_t)colB * FIN_ + kb);
        cq = WMMA_BF16(a, bq, cq);
        ck = WMMA_BF16(a, bk, ck);
        cv = WMMA_BF16(a, bv, cv);
        cr = WMMA_BF16(a, br, cr);
    }
    int rbase = m0 + hs * 8;          // 8 consecutive output rows (same batch; 512%16==0)
    int col   = n0 + (lane & 15);
    float bsv = bs[col];
    V8B kt, vt;
#pragma unroll
    for (int r = 0; r < 8; ++r) {
        size_t row = (size_t)(rbase + r);
        float q = cq[r]; q = (q > 0.f) ? q + 1.0f : __expf(q);   // elu+1
        float k = ck[r]; k = (k > 0.f) ? k + 1.0f : __expf(k);
        float v = cv[r];
        Qf[row * D_ + col] = q;
        Kf[row * D_ + col] = k;
        Vf[row * D_ + col] = v;
        Rf[row * D_ + col] = cr[r] + bsv;
        kt.e[r] = (__bf16)k;
        vt.e[r] = (__bf16)v;
    }
    int b  = rbase >> 9;              // /T_
    int t0 = rbase & (T_ - 1);
    *(v8bf*)(Ktb + ((size_t)(b * D_ + col)) * T_ + t0) = kt.v;   // [B][D][T]
    *(v8bf*)(Vtb + ((size_t)(b * D_ + col)) * T_ + t0) = vt.v;
}

// ============ 3a. per-chunk outer-product totals  Csum[b][g] = K_g^T V_g  (WMMA) ============
__global__ void la_chunk_sums(const __bf16* __restrict__ Ktb,
                              const __bf16* __restrict__ Vtb,
                              float* __restrict__ Csum) {
    int lane = threadIdx.x & 31;
    int tile = blockIdx.x * 4 + (threadIdx.x >> 5);   // 0..4095
    int bg  = tile >> 6;                               // batch*NCH+chunk
    int sub = tile & 63;                               // 8x8 D-tiles
    int b = bg >> 3, g = bg & 7;
    int m0 = (sub >> 3) * 16, n0 = (sub & 7) * 16;
    int hs = lane >> 4;
    int mm = m0 + (lane & 15), nn = n0 + (lane & 15);
    v8f acc = {};
#pragma unroll
    for (int kk = 0; kk < CHUNK; kk += 32) {
        v16bf a  = ldA(Ktb + ((size_t)(b * D_ + mm)) * T_ + g * CHUNK + kk + hs * 16);
        v16bf bb = ldB(Vtb + ((size_t)(b * D_ + nn)) * T_ + g * CHUNK + kk + hs * 16);
        acc = WMMA_BF16(a, bb, acc);
    }
    float* out = Csum + (size_t)bg * DD;
#pragma unroll
    for (int r = 0; r < 8; ++r)
        out[(size_t)(m0 + hs * 8 + r) * D_ + n0 + (lane & 15)] = acc[r];
}

// ============ 3b. exclusive prefix over chunks ============
__global__ void la_chunk_prefix(const float* __restrict__ Csum, float* __restrict__ Sstart) {
    int b = blockIdx.y;
    int p = blockIdx.x * 256 + threadIdx.x;   // 0..16383
    float acc = 0.f;
#pragma unroll
    for (int g = 0; g < NCH; ++g) {
        size_t o = (size_t)(b * NCH + g) * DD + p;
        Sstart[o] = acc;
        acc += Csum[o];
    }
}

// ============ 3c. intra-chunk scan: write S_out (b128 NT stores) + numerator Q.S ============
// Grid (coltile=4, chunk=8, batch=8) x 256 threads.
// Thread tile: 4 rows x 4 consecutive cols -> each step emits 4x global_store_b128 (NT),
// i.e. 512B per wave store instruction into the 268MB S_out stream.
// Thread t: cg = t&7 (cols jt*32 + cg*4 .. +3), rg = t>>3 (rows rg*4 .. +3).
__global__ void la_scan_chunk(const float* __restrict__ Qf, const float* __restrict__ Kf,
                              const float* __restrict__ Vf,
                              const float* __restrict__ Sstart, const float* __restrict__ S0,
                              float* __restrict__ S_out, float* __restrict__ numf) {
    __shared__ float Kl[D_], Ql[D_], Vl[32], red[256];
    int jt = blockIdx.x, g = blockIdx.y, b = blockIdx.z;
    int tid = threadIdx.x;
    int cg = tid & 7;                 // col group (4 cols)
    int rg = tid >> 3;                // row group (4 rows), 0..31
    int wv = tid >> 5;                // wave id 0..7
    int col0 = jt * 32 + cg * 4;
    int r0 = rg * 4;
    float s[4][4], s0v[4][4];
    const float* ss  = Sstart + (size_t)(b * NCH + g) * DD;
    const float* s0p = S0 + (size_t)b * DD;
#pragma unroll
    for (int r = 0; r < 4; ++r)
#pragma unroll
        for (int cc = 0; cc < 4; ++cc) {
            s[r][cc]   = ss[(size_t)(r0 + r) * D_ + col0 + cc];
            s0v[r][cc] = s0p[(size_t)(r0 + r) * D_ + col0 + cc];
        }
    for (int t = 0; t < CHUNK; ++t) {
        size_t row = (size_t)b * T_ + g * CHUNK + t;
        if (tid < D_) Kl[tid] = Kf[row * D_ + tid];
        else          Ql[tid - D_] = Qf[row * D_ + tid - D_];
        if (tid < 32) Vl[tid] = Vf[row * D_ + jt * 32 + tid];
        __syncthreads();
        float vv0 = Vl[cg * 4 + 0], vv1 = Vl[cg * 4 + 1];
        float vv2 = Vl[cg * 4 + 2], vv3 = Vl[cg * 4 + 3];
        v4f part = {0.f, 0.f, 0.f, 0.f};
        float* sbase = S_out + row * (size_t)DD + col0;
#pragma unroll
        for (int r = 0; r < 4; ++r) {
            int i = r0 + r;
            float k = Kl[i], q = Ql[i];
            s[r][0] = fmaf(k, vv0, s[r][0]);
            s[r][1] = fmaf(k, vv1, s[r][1]);
            s[r][2] = fmaf(k, vv2, s[r][2]);
            s[r][3] = fmaf(k, vv3, s[r][3]);
            v4f pack;
            pack.x = clip1e20(s[r][0]) + s0v[r][0];
            pack.y = clip1e20(s[r][1]) + s0v[r][1];
            pack.z = clip1e20(s[r][2]) + s0v[r][2];
            pack.w = clip1e20(s[r][3]) + s0v[r][3];
            __builtin_nontemporal_store(pack, (v4f*)(sbase + (size_t)i * D_));
            part.x = fmaf(q, pack.x, part.x);
            part.y = fmaf(q, pack.y, part.y);
            part.z = fmaf(q, pack.z, part.z);
            part.w = fmaf(q, pack.w, part.w);
        }
        // in-wave reduce over the wave's 4 row-groups (lanes l, l^8, l^16 share (cg,cc))
#pragma unroll
        for (int m = 8; m <= 16; m <<= 1) {
            part.x += __shfl_xor(part.x, m, 32);
            part.y += __shfl_xor(part.y, m, 32);
            part.z += __shfl_xor(part.z, m, 32);
            part.w += __shfl_xor(part.w, m, 32);
        }
        if ((tid & 31) < 8)
            *(v4f*)&red[wv * 32 + cg * 4] = part;   // one ds_store_b128 per owning lane
        __syncthreads();
        if (tid < 32) {
            float n = red[tid];
#pragma unroll
            for (int w = 1; w < 8; ++w) n += red[w * 32 + tid];
            numf[row * D_ + jt * 32 + tid] = n;
        }
        __syncthreads();
    }
}

// ============ 4. Z cumsum ============
__global__ void la_z_scan(const float* __restrict__ Kf, const float* __restrict__ Z0,
                          float* __restrict__ Zout) {
    int b = blockIdx.x, i = threadIdx.x;   // 128 threads
    float z  = 0.f;
    float z0 = Z0[b * D_ + i];
    for (int t = 0; t < T_; ++t) {
        size_t o = ((size_t)b * T_ + t) * D_ + i;
        z += Kf[o];
        Zout[o] = clip1e20(z) + z0;
    }
}

// ============ 5. denominator Q.Z + h = num/den (bf16 for FFN GEMM) ============
__global__ void la_den_h(const float* __restrict__ Qf, const float* __restrict__ Zout,
                         const float* __restrict__ numf, __bf16* __restrict__ hb) {
    int row  = blockIdx.x * 8 + (threadIdx.x >> 5);
    int lane = threadIdx.x & 31;
    const float* q = Qf   + (size_t)row * D_;
    const float* z = Zout + (size_t)row * D_;
    float acc = 0.f;
#pragma unroll
    for (int i = 0; i < 4; ++i) acc += q[lane * 4 + i] * z[lane * 4 + i];
#pragma unroll
    for (int m = 16; m >= 1; m >>= 1) acc += __shfl_xor(acc, m, 32);
    float inv = 1.0f / (acc + 1e-5f);
    const float* n = numf + (size_t)row * D_;
    V4B o;
#pragma unroll
    for (int i = 0; i < 4; ++i) o.e[i] = (__bf16)(n[lane * 4 + i] * inv);
    *(v4bf*)(hb + (size_t)row * D_ + lane * 4) = o.v;
}

// ============ 6. FFN GEMM [4096,128]x[128,128] (WMMA) with bias/relu/residual epilogue ============
__global__ void la_ffn_gemm(const __bf16* __restrict__ A, const __bf16* __restrict__ Wt,
                            const float* __restrict__ bias, const float* __restrict__ Radd,
                            float* __restrict__ outf, __bf16* __restrict__ outb) {
    int lane = threadIdx.x & 31;
    int tile = blockIdx.x * 4 + (threadIdx.x >> 5);   // 0..2047
    int m0 = (tile >> 3) * 16, n0 = (tile & 7) * 16;
    int hs = lane >> 4;
    int rowA = m0 + (lane & 15);
    int colB = n0 + (lane & 15);
    v8f acc = {};
#pragma unroll
    for (int kk = 0; kk < D_; kk += 32) {
        v16bf a = ldA(A  + (size_t)rowA * D_ + kk + hs * 16);
        v16bf b = ldB(Wt + (size_t)colB * D_ + kk + hs * 16);
        acc = WMMA_BF16(a, b, acc);
    }
    int rbase = m0 + hs * 8;
    int col   = n0 + (lane & 15);
    float bv = bias[col];
#pragma unroll
    for (int r = 0; r < 8; ++r) {
        size_t row = (size_t)(rbase + r);
        float o = fmaxf(acc[r] + bv, 0.f);                 // relu
        if (Radd) o += Radd[row * D_ + col];               // + xn@Ws + bs
        if (outf) outf[row * D_ + col] = o;
        if (outb) outb[row * D_ + col] = (__bf16)o;
    }
}

// ================================ launcher ================================
extern "C" void kernel_launch(void* const* d_in, const int* in_sizes, int n_in,
                              void* d_out, int out_size, void* d_ws, size_t ws_size,
                              hipStream_t stream) {
    // inputs: x S0 Z0 gamma beta Wk Wq Wv W1 b1 W2 b2 Ws bs
    const float* x     = (const float*)d_in[0];
    const float* S0    = (const float*)d_in[1];
    const float* Z0    = (const float*)d_in[2];
    const float* gamma = (const float*)d_in[3];
    const float* beta  = (const float*)d_in[4];
    const float* Wk    = (const float*)d_in[5];
    const float* Wq    = (const float*)d_in[6];
    const float* Wv    = (const float*)d_in[7];
    const float* W1    = (const float*)d_in[8];
    const float* b1    = (const float*)d_in[9];
    const float* W2    = (const float*)d_in[10];
    const float* b2    = (const float*)d_in[11];
    const float* Ws    = (const float*)d_in[12];
    const float* bs    = (const float*)d_in[13];

    // d_out = [out (B*T*D) | S_out (B*T*D*D) | Z (B*T*D)] fp32
    float* out_p = (float*)d_out;
    float* Sout  = out_p + (size_t)BT_ * D_;
    float* Zout  = Sout  + (size_t)BT_ * DD;

    // workspace carve-up (~26 MB needed)
    char* base = (char*)d_ws;
    size_t off = 0;
    auto carve = [&](size_t bytes) -> char* {
        char* p = base + off;
        off = (off + bytes + 255) & ~(size_t)255;
        return p;
    };
    __bf16* xnb  = (__bf16*)carve((size_t)BT_ * FIN_ * 2);
    __bf16* Wkt  = (__bf16*)carve((size_t)FIN_ * D_ * 2);
    __bf16* Wqt  = (__bf16*)carve((size_t)FIN_ * D_ * 2);
    __bf16* Wvt  = (__bf16*)carve((size_t)FIN_ * D_ * 2);
    __bf16* Wst  = (__bf16*)carve((size_t)FIN_ * D_ * 2);
    __bf16* W1t  = (__bf16*)carve((size_t)D_ * D_ * 2);
    __bf16* W2t  = (__bf16*)carve((size_t)D_ * D_ * 2);
    float*  Qf   = (float*) carve((size_t)BT_ * D_ * 4);
    float*  Kf   = (float*) carve((size_t)BT_ * D_ * 4);
    float*  Vf   = (float*) carve((size_t)BT_ * D_ * 4);
    float*  Rf   = (float*) carve((size_t)BT_ * D_ * 4);
    __bf16* Ktb  = (__bf16*)carve((size_t)B_ * D_ * T_ * 2);
    __bf16* Vtb  = (__bf16*)carve((size_t)B_ * D_ * T_ * 2);
    float*  Csum = (float*) carve((size_t)B_ * NCH * DD * 4);
    float*  Sst  = (float*) carve((size_t)B_ * NCH * DD * 4);
    float*  numf = (float*) carve((size_t)BT_ * D_ * 4);
    __bf16* hb   = (__bf16*)carve((size_t)BT_ * D_ * 2);
    __bf16* h1b  = (__bf16*)carve((size_t)BT_ * D_ * 2);
    (void)ws_size; (void)in_sizes; (void)n_in; (void)out_size;

    la_prep_weights<<<dim3(128, 6), 256, 0, stream>>>(Wk, Wq, Wv, W1, W2, Ws,
                                                      Wkt, Wqt, Wvt, W1t, W2t, Wst);
    la_layernorm<<<BT_ / 8, 256, 0, stream>>>(x, gamma, beta, xnb);
    la_qkvr_gemm<<<512, 128, 0, stream>>>(xnb, Wkt, Wqt, Wvt, Wst, bs,
                                          Qf, Kf, Vf, Rf, Ktb, Vtb);
    la_chunk_sums<<<1024, 128, 0, stream>>>(Ktb, Vtb, Csum);
    la_chunk_prefix<<<dim3(64, B_), 256, 0, stream>>>(Csum, Sst);
    la_scan_chunk<<<dim3(4, NCH, B_), 256, 0, stream>>>(Qf, Kf, Vf, Sst, S0, Sout, numf);
    la_z_scan<<<B_, D_, 0, stream>>>(Kf, Z0, Zout);
    la_den_h<<<BT_ / 8, 256, 0, stream>>>(Qf, Zout, numf, hb);
    la_ffn_gemm<<<512, 128, 0, stream>>>(hb, W1t, b1, nullptr, nullptr, h1b);
    la_ffn_gemm<<<512, 128, 0, stream>>>(h1b, W2t, b2, Rf, out_p, nullptr);
}
